// Global_Critic_22419729285739
// MI455X (gfx1250) — matrix operand order repaired and checked by reading, not compiled, not run on hardware
//
#include <hip/hip_runtime.h>
#include <stdint.h>

typedef __attribute__((ext_vector_type(16))) __bf16 v16bf;
typedef __attribute__((ext_vector_type(8)))  float  v8f;

#define NAG   8
#define BATCH 8192
#define DOBS  128
#define DACT  32
#define DOA   160     // OBS + ACT
#define EDIM  128
#define DH    1152    // (N+1)*E
#define BT    64      // batch rows per workgroup
#define LSTR  136     // padded LDS row stride (bf16 elems) -> conflict-free b128 loads

static __device__ __forceinline__ __bf16 f2bf(float f) { return (__bf16)f; }
static __device__ __forceinline__ float bf2f(__bf16 h) { return (float)h; }

union Frag {
  v16bf v;
  uint4 q[2];
  __bf16 h[16];
};

static __device__ __forceinline__ v8f wmma_bf16(v16bf a, v16bf b, v8f c) {
  // D = A(16x32 bf16) x B(32x16 bf16) + C(16x16 f32)
  return __builtin_amdgcn_wmma_f32_16x16x32_bf16(false, a, false, b, (short)0, c, false, false);
}

// A fragment (16x32) from LDS bf16, row-major with stride LSTR.
// ISA A layout: lane row = lane&15; lanes 0-15 hold K chunks [0..7]+[16..23], lanes 16-31 [8..15]+[24..31].
static __device__ __forceinline__ v16bf ldsA(const __bf16* base, int row0, int kb, int lane) {
  int row = row0 + (lane & 15);
  int kc  = kb + ((lane >> 4) << 3);
  const __bf16* p = base + row * LSTR + kc;
  Frag f;
  f.q[0] = *(const uint4*)(p);
  f.q[1] = *(const uint4*)(p + 16);
  return f.v;
}

// B fragment (32x16) from global bf16 transposed weights WT[out][in], stride ldw.
// ISA B layout: column = lane&15; lanes 0-15 hold K chunks [0..7]+[16..23], lanes 16-31 [8..15]+[24..31].
static __device__ __forceinline__ v16bf gldB(const __bf16* wt, int ldw, int col0, int kb, int lane) {
  int n = col0 + (lane & 15);
  int k = kb + ((lane >> 4) << 3);
  const __bf16* p = wt + (size_t)n * ldw + k;
  Frag f;
  f.q[0] = ((const uint4*)p)[0];
  f.q[1] = ((const uint4*)p)[2];
  return f.v;
}

// A fragment from global f32 oa = [obs | act] of agent j, converted to bf16 on the fly.
static __device__ __forceinline__ v16bf gldA_oa(const float* obs, const float* act,
                                                int j, int row0, int kb, int lane) {
  int b  = row0 + (lane & 15);
  int c0 = kb + ((lane >> 4) << 3);
  Frag f;
#pragma unroll
  for (int half = 0; half < 2; ++half) {
    int c = c0 + half * 16;
    const float* src = (c < DOBS)
        ? obs + ((size_t)j * BATCH + b) * DOBS + c
        : act + ((size_t)j * BATCH + b) * DACT + (c - DOBS);
    float4 p0 = ((const float4*)src)[0];
    float4 p1 = ((const float4*)src)[1];
    int o = half * 8;
    f.h[o+0] = f2bf(p0.x); f.h[o+1] = f2bf(p0.y);
    f.h[o+2] = f2bf(p0.z); f.h[o+3] = f2bf(p0.w);
    f.h[o+4] = f2bf(p1.x); f.h[o+5] = f2bf(p1.y);
    f.h[o+6] = f2bf(p1.z); f.h[o+7] = f2bf(p1.w);
  }
  return f.v;
}

// ---------------------------------------------------------------------------
// Prep: convert + transpose weights f32 -> bf16 [out][in] so B-fragment loads
// are two 16B reads/lane along k (served from the 192MB L2 after first touch).
// ---------------------------------------------------------------------------
__global__ void prep_weights(const float* __restrict__ Wg, const float* __restrict__ Wv,
                             const float* __restrict__ W1,
                             __bf16* __restrict__ WgT, __bf16* __restrict__ WvT,
                             __bf16* __restrict__ W1T) {
  int t = blockIdx.x * blockDim.x + threadIdx.x;
  const int T0 = NAG * EDIM * DOA;
  const int T1 = NAG * EDIM * EDIM;
  const int T2 = NAG * EDIM * DH;
  if (t < T0) {
    int n = t / (EDIM * DOA), rem = t % (EDIM * DOA);
    int o = rem / DOA, d = rem % DOA;
    WgT[t] = f2bf(Wg[((size_t)n * DOA + d) * EDIM + o]);
  } else if (t < T0 + T1) {
    int u = t - T0;
    int n = u / (EDIM * EDIM), rem = u % (EDIM * EDIM);
    int o = rem / EDIM, d = rem % EDIM;
    WvT[u] = f2bf(Wv[((size_t)n * EDIM + d) * EDIM + o]);
  } else if (t < T0 + T1 + T2) {
    int u = t - T0 - T1;
    int n = u / (EDIM * DH), rem = u % (EDIM * DH);
    int o = rem / DH, d = rem % DH;
    W1T[u] = f2bf(W1[((size_t)n * DH + d) * EDIM + o]);
  }
}

// ---------------------------------------------------------------------------
// Fully fused critic: one WG = one 64-row batch tile, all agents & heads.
// 256 threads = 8 wave32. Wave w owns output tile rows (w&3)*16.., cols (w>>2)*64..
// LDS-limited to 1 WG/WGP (2 waves/SIMD) -> spend VGPRs on operand reuse.
// ---------------------------------------------------------------------------
__global__ __launch_bounds__(256, 1) void critic_fused(
    const float* __restrict__ obs, const float* __restrict__ act,
    const __bf16* __restrict__ WgT, const float* __restrict__ bg,
    const __bf16* __restrict__ WvT, const float* __restrict__ bv,
    const __bf16* __restrict__ W1T, const float* __restrict__ b1,
    const float* __restrict__ W2, const float* __restrict__ b2,
    float* __restrict__ out)
{
  __shared__ __align__(16) __bf16 e_lds[NAG][BT * LSTR];  // e_j, bf16   (136 KB)
  __shared__ __align__(16) __bf16 S_lds[NAG][BT * LSTR];  // S_l, bf16   (136 KB)
  __shared__ __align__(16) float h_lds[BT * LSTR];        // h (f32) / t (bf16) (34 KB)
  __bf16* t_lds = (__bf16*)h_lds;

  const int tid  = threadIdx.x;
  const int lane = tid & 31;
  const int w    = tid >> 5;
  const int rB   = (w & 3) * 16;       // row-tile base
  const int cH   = (w >> 2) * 64;      // column-half base
  const int b0   = blockIdx.x * BT;
  const int hiq  = (lane >> 4) << 3;   // C-frag: upper lanes hold rows +8
  const int colL = lane & 15;

  // -------- Phase 0: e_j = relu(oa_j @ Wg_j + bg_j), all 8 agents ----------
  for (int j = 0; j < NAG; ++j) {
    v8f acc[4];
#pragma unroll
    for (int ct = 0; ct < 4; ++ct)
#pragma unroll
      for (int g = 0; g < 8; ++g) acc[ct][g] = 0.f;
    for (int ks = 0; ks < 5; ++ks) {          // K = 160
      int kb = ks * 32;
      v16bf a = gldA_oa(obs, act, j, b0 + rB, kb, lane);
#pragma unroll
      for (int ct = 0; ct < 4; ++ct) {
        v16bf bm = gldB(WgT + (size_t)j * EDIM * DOA, DOA, cH + ct * 16, kb, lane);
        acc[ct] = wmma_bf16(a, bm, acc[ct]);
      }
    }
#pragma unroll
    for (int ct = 0; ct < 4; ++ct) {
      int col = cH + ct * 16 + colL;
      float bgv = bg[j * EDIM + col];
#pragma unroll
      for (int g = 0; g < 8; ++g) {
        float vv = acc[ct][g] + bgv;
        vv = vv > 0.f ? vv : 0.f;
        e_lds[j][(rB + hiq + g) * LSTR + col] = f2bf(vv);
      }
    }
  }
  __syncthreads();

  // -------- Phase 1: S_l = sum_j leaky_relu(e_j @ Wv_l + bv_l) -------------
  // 4 j's per pass: each Wv B-fragment load feeds 4 WMMAs.
  for (int l = 0; l < NAG; ++l) {
    v8f sacc[4];
#pragma unroll
    for (int ct = 0; ct < 4; ++ct)
#pragma unroll
      for (int g = 0; g < 8; ++g) sacc[ct][g] = 0.f;
    for (int jq = 0; jq < NAG; jq += 4) {
      v8f c[4][4];                       // [jj][ct]
#pragma unroll
      for (int jj = 0; jj < 4; ++jj)
#pragma unroll
        for (int ct = 0; ct < 4; ++ct)
#pragma unroll
          for (int g = 0; g < 8; ++g) c[jj][ct][g] = 0.f;
      for (int kq = 0; kq < 4; ++kq) {
        int kb = kq * 32;
        v16bf a0 = ldsA(&e_lds[jq + 0][0], rB, kb, lane);
        v16bf a1 = ldsA(&e_lds[jq + 1][0], rB, kb, lane);
        v16bf a2 = ldsA(&e_lds[jq + 2][0], rB, kb, lane);
        v16bf a3 = ldsA(&e_lds[jq + 3][0], rB, kb, lane);
#pragma unroll
        for (int ct = 0; ct < 4; ++ct) {
          v16bf bm = gldB(WvT + (size_t)l * EDIM * EDIM, EDIM, cH + ct * 16, kb, lane);
          c[0][ct] = wmma_bf16(a0, bm, c[0][ct]);
          c[1][ct] = wmma_bf16(a1, bm, c[1][ct]);
          c[2][ct] = wmma_bf16(a2, bm, c[2][ct]);
          c[3][ct] = wmma_bf16(a3, bm, c[3][ct]);
        }
      }
#pragma unroll
      for (int jj = 0; jj < 4; ++jj)
#pragma unroll
        for (int ct = 0; ct < 4; ++ct) {
          float bvv = bv[l * EDIM + cH + ct * 16 + colL];
#pragma unroll
          for (int g = 0; g < 8; ++g) {
            float vv = c[jj][ct][g] + bvv;
            vv = vv > 0.f ? vv : 0.01f * vv;   // leaky_relu, slope 0.01
            sacc[ct][g] += vv;
          }
        }
    }
#pragma unroll
    for (int ct = 0; ct < 4; ++ct) {
      int col = cH + ct * 16 + colL;
#pragma unroll
      for (int g = 0; g < 8; ++g)
        S_lds[l][(rB + hiq + g) * LSTR + col] = f2bf(sacc[ct][g]);
    }
  }
  __syncthreads();

  // -------- Phase 2: per output agent i ------------------------------------
  for (int i = 0; i < NAG; ++i) {
    // e_i A-fragments are invariant over the l-loop: hoist into registers.
    v16bf aE[4];
#pragma unroll
    for (int kq = 0; kq < 4; ++kq)
      aE[kq] = ldsA(&e_lds[i][0], rB, kq * 32, lane);

    v8f hacc[4];
#pragma unroll
    for (int ct = 0; ct < 4; ++ct)
#pragma unroll
      for (int g = 0; g < 8; ++g) hacc[ct][g] = 0.f;

    for (int l = 0; l < NAG; ++l) {
      // recompute v[l,i] tile; t = S_l - v[l,i]  (== sum_{j!=i} v[l,j])
      v8f c[4];
#pragma unroll
      for (int ct = 0; ct < 4; ++ct)
#pragma unroll
        for (int g = 0; g < 8; ++g) c[ct][g] = 0.f;
#pragma unroll
      for (int kq = 0; kq < 4; ++kq) {
        int kb = kq * 32;
#pragma unroll
        for (int ct = 0; ct < 4; ++ct)
          c[ct] = wmma_bf16(aE[kq], gldB(WvT + (size_t)l * EDIM * EDIM, EDIM, cH + ct * 16, kb, lane), c[ct]);
      }
#pragma unroll
      for (int ct = 0; ct < 4; ++ct) {
        int col = cH + ct * 16 + colL;
        float bvv = bv[l * EDIM + col];
#pragma unroll
        for (int g = 0; g < 8; ++g) {
          int row = rB + hiq + g;
          float vv = c[ct][g] + bvv;
          vv = vv > 0.f ? vv : 0.01f * vv;
          float sv = bf2f(S_lds[l][row * LSTR + col]);
          t_lds[row * LSTR + col] = f2bf(sv - vv);
        }
      }
      __syncthreads();
      // hacc += t @ W1_i[block l]
      for (int kq = 0; kq < 4; ++kq) {
        int kb = kq * 32;
        v16bf a = ldsA(t_lds, rB, kb, lane);
#pragma unroll
        for (int ct = 0; ct < 4; ++ct)
          hacc[ct] = wmma_bf16(a, gldB(W1T + (size_t)i * EDIM * DH, DH, cH + ct * 16, l * EDIM + kb, lane), hacc[ct]);
      }
      __syncthreads();
    }
    // hacc += e_i @ W1_i[last 128 input rows]
#pragma unroll
    for (int kq = 0; kq < 4; ++kq) {
      int kb = kq * 32;
#pragma unroll
      for (int ct = 0; ct < 4; ++ct)
        hacc[ct] = wmma_bf16(aE[kq], gldB(W1T + (size_t)i * EDIM * DH, DH, cH + ct * 16, NAG * EDIM + kb, lane), hacc[ct]);
    }
    // bias + relu, stage h as f32 in LDS
#pragma unroll
    for (int ct = 0; ct < 4; ++ct) {
      int col = cH + ct * 16 + colL;
      float b1v = b1[i * EDIM + col];
#pragma unroll
      for (int g = 0; g < 8; ++g) {
        float hv = hacc[ct][g] + b1v;
        hv = hv > 0.f ? hv : 0.f;
        h_lds[(rB + hiq + g) * LSTR + col] = hv;
      }
    }
    __syncthreads();
    // out[i, b] = h . W2_i + b2_i   (one thread per batch row)
    if (tid < BT) {
      float sum = b2[i];
      const float* w2 = W2 + i * EDIM;
      for (int e = 0; e < EDIM; ++e)
        sum += h_lds[tid * LSTR + e] * w2[e];
      out[(size_t)i * BATCH + b0 + tid] = sum;
    }
    __syncthreads();
  }
}

extern "C" void kernel_launch(void* const* d_in, const int* in_sizes, int n_in,
                              void* d_out, int out_size, void* d_ws, size_t ws_size,
                              hipStream_t stream) {
  (void)in_sizes; (void)n_in; (void)out_size; (void)ws_size;
  const float* obs = (const float*)d_in[0];
  const float* act = (const float*)d_in[1];
  const float* Wg  = (const float*)d_in[2];
  const float* bg  = (const float*)d_in[3];
  // d_in[4..7] = Wq, bq, Wk, bk : dead code (softmax over singleton axis == 1)
  const float* Wv  = (const float*)d_in[8];
  const float* bv  = (const float*)d_in[9];
  const float* W1  = (const float*)d_in[10];
  const float* b1  = (const float*)d_in[11];
  const float* W2  = (const float*)d_in[12];
  const float* b2  = (const float*)d_in[13];
  float* out = (float*)d_out;

  __bf16* WgT = (__bf16*)d_ws;                          // [8][128][160] bf16
  __bf16* WvT = WgT + (size_t)NAG * EDIM * DOA;         // [8][128][128] bf16
  __bf16* W1T = WvT + (size_t)NAG * EDIM * EDIM;        // [8][128][1152] bf16
  // total scratch: ~2.95 MB

  int totalPrep = NAG * EDIM * (DOA + EDIM + DH);
  prep_weights<<<(totalPrep + 255) / 256, 256, 0, stream>>>(Wg, Wv, W1, WgT, WvT, W1T);
  critic_fused<<<BATCH / BT, 256, 0, stream>>>(obs, act, WgT, bg, WvT, bv, W1T, b1, W2, b2, out);
}
